// GAT_8546984919351
// MI455X (gfx1250) — matrix operand order repaired and checked by reading, not compiled, not run on hardware
//
#include <hip/hip_runtime.h>
#include <cstdint>
#include <cstddef>

// ---------------------------------------------------------------------------
// GAT 5-layer pipeline for MI455X (gfx1250, wave32, WMMA).
// GEMMs via v_wmma_f32_16x16x32_bf16 with per-wave A-fragment reuse across
// all column tiles; graph softmax/aggregation via f32 atomics (whole working
// set is L2-resident on the 192MB L2).
// ---------------------------------------------------------------------------

typedef __attribute__((ext_vector_type(16))) __bf16 v16bf;
typedef __attribute__((ext_vector_type(8)))  float  v8f;

// ---------------------------- small utilities ------------------------------

__global__ void fill_f32(float* __restrict__ p, float v, int n) {
  int i = blockIdx.x * blockDim.x + threadIdx.x;
  if (i < n) p[i] = v;
}

// edge_index [2,E] int32 -> src/dst int32 with N appended self loops
__global__ void build_edges(const int* __restrict__ ei, int E, int N,
                            int* __restrict__ src, int* __restrict__ dst) {
  int i = blockIdx.x * blockDim.x + threadIdx.x;
  int Et = E + N;
  if (i >= Et) return;
  if (i < E) { src[i] = ei[i]; dst[i] = ei[E + i]; }
  else       { src[i] = i - E; dst[i] = i - E; }
}

// f32 activation [rows, width] (leading dim ld_in) -> bf16 [rows, Kpad], zero pad
__global__ void to_bf16_pad(const float* __restrict__ X, int ld_in, int width,
                            int rows, __bf16* __restrict__ Xb, int Kpad) {
  int i = blockIdx.x * blockDim.x + threadIdx.x;
  if (i >= rows * Kpad) return;
  int r = i / Kpad, k = i - r * Kpad;
  float v = (k < width) ? X[(size_t)r * ld_in + k] : 0.0f;
  Xb[i] = (__bf16)v;
}

// W [Fin, Fout] f32 -> bf16 [Kpad, Npad], zero pad
__global__ void w_to_bf16(const float* __restrict__ W, int Fin, int Fout,
                          __bf16* __restrict__ Wb, int Kpad, int Npad) {
  int i = blockIdx.x * blockDim.x + threadIdx.x;
  if (i >= Kpad * Npad) return;
  int k = i / Npad, n = i - k * Npad;
  float v = (k < Fin && n < Fout) ? W[(size_t)k * Fout + n] : 0.0f;
  Wb[i] = (__bf16)v;
}

// ------------------------------- WMMA GEMM ---------------------------------
// C[M, Npad] = A[M, Kpad](bf16) x B[Kpad, Npad](bf16), f32 accumulate.
// One wave owns a 16-row stripe and all NT 16-wide column tiles:
// A fragment loaded once per K-step, NT WMMAs issued back-to-back against
// pre-clustered B fragment loads.
template <int NT>
__global__ void gemm_wmma_bf16_nt(const __bf16* __restrict__ A,
                                  const __bf16* __restrict__ B,
                                  float* __restrict__ C,
                                  int Mtiles, int Kpad, int Npad, int M) {
  int wid  = (blockIdx.x * blockDim.x + threadIdx.x) >> 5;
  int lane = threadIdx.x & 31;
  if (wid >= Mtiles) return;
  int half = lane >> 4;        // lanes 16-31 -> second K/M half per ISA layout
  int mrow = lane & 15;

  int arow = wid * 16 + mrow;
  if (arow >= M) arow = M - 1;               // safety (M divisible by 16 here)
  const __bf16* __restrict__ Ar = A + (size_t)arow * Kpad;

  v8f acc[NT];
  v8f zero = {};
#pragma unroll
  for (int t = 0; t < NT; ++t) acc[t] = zero;

  for (int k0 = 0; k0 < Kpad; k0 += 32) {
    // A fragment: 16-bit A 16x32 layout (ISA 7.12.2); lowers to 2x b128 loads
    v16bf af;
#pragma unroll
    for (int j = 0; j < 8; ++j) {
      int k = ((j >= 4) ? 16 : 0) + half * 8 + (j & 3) * 2;
      af[2 * j]     = Ar[k0 + k];
      af[2 * j + 1] = Ar[k0 + k + 1];
    }
    // B fragments: lane holds B column N = t*16+(lane&15), elements over K in same order as A
    const __bf16* __restrict__ Bcol = B + (size_t)k0 * Npad + (lane & 15);
    v16bf bfv[NT];
#pragma unroll
    for (int t = 0; t < NT; ++t)
#pragma unroll
      for (int s = 0; s < 16; ++s)
        bfv[t][s] = Bcol[(size_t)(((s >= 8) ? 16 : 0) + half * 8 + (s & 7)) * Npad + t * 16];
#pragma unroll
    for (int t = 0; t < NT; ++t)
      acc[t] = __builtin_amdgcn_wmma_f32_16x16x32_bf16(
          false, af, false, bfv[t], (short)0, acc[t], false, false);
  }
  // C/D layout: VGPR r -> M = r + 8*half, N = lane&15
#pragma unroll
  for (int t = 0; t < NT; ++t) {
    int ncol = t * 16 + (lane & 15);
#pragma unroll
    for (int r = 0; r < 8; ++r) {
      int m = wid * 16 + r + 8 * half;
      if (m < M) C[(size_t)m * Npad + ncol] = acc[t][r];
    }
  }
}

// --------------------------- attention pipeline ----------------------------

// e_src[n,h] = sum_c H[n, h*C+c]*a_src[h,c]; same for e_dst
__global__ void attn_scores(const float* __restrict__ Hb, int Npad,
                            const float* __restrict__ as_, const float* __restrict__ ad_,
                            float* __restrict__ es, float* __restrict__ ed,
                            int Nn, int Hh, int Cc) {
  int i = blockIdx.x * blockDim.x + threadIdx.x;
  if (i >= Nn * Hh) return;
  int n = i / Hh, h = i - n * Hh;
  const float* hp  = Hb + (size_t)n * Npad + h * Cc;
  const float* asp = as_ + h * Cc;
  const float* adp = ad_ + h * Cc;
  float s0 = 0.f, s1 = 0.f;
  for (int c = 0; c < Cc; ++c) { float v = hp[c]; s0 += v * asp[c]; s1 += v * adp[c]; }
  es[i] = s0; ed[i] = s1;
}

__device__ __forceinline__ void atomicMaxF(float* addr, float val) {
  __hip_atomic_fetch_max(addr, val, __ATOMIC_RELAXED, __HIP_MEMORY_SCOPE_AGENT);
}

// pass 1: e = leaky_relu(es[src]+ed[dst], 0.2); store; segment max over dst
__global__ void edge_max(const int* __restrict__ src, const int* __restrict__ dst,
                         const float* __restrict__ es, const float* __restrict__ ed,
                         float* __restrict__ ebuf, float* __restrict__ emax,
                         int Et, int Hh) {
  int i = blockIdx.x * blockDim.x + threadIdx.x;
  if (i >= Et * Hh) return;
  int e = i / Hh, h = i - e * Hh;
  int s = src[e], d = dst[e];
  float v = es[s * Hh + h] + ed[d * Hh + h];
  v = (v > 0.f) ? v : 0.2f * v;
  ebuf[i] = v;
  atomicMaxF(&emax[d * Hh + h], v);
}

// pass 2: ex = exp(e - emax[dst]); store; segment sum -> denom
__global__ void edge_exp(const int* __restrict__ dst, float* __restrict__ ebuf,
                         const float* __restrict__ emax, float* __restrict__ denom,
                         int Et, int Hh) {
  int i = blockIdx.x * blockDim.x + threadIdx.x;
  if (i >= Et * Hh) return;
  int e = i / Hh, h = i - e * Hh;
  int d = dst[e];
  float x = __expf(ebuf[i] - emax[d * Hh + h]);
  ebuf[i] = x;
  atomicAdd(&denom[d * Hh + h], x);
}

// pass 3: out[dst] += H[src] * alpha. Flat thread-per-(edge, channel) with
// compile-time H,C so div/mod become constant-multiplier sequences; adjacent
// lanes cover adjacent channels of one edge (coalesced gather + atomics).
template <int HH, int CC>
__global__ void edge_aggregate_t(const int* __restrict__ src, const int* __restrict__ dst,
                                 const float* __restrict__ ebuf, const float* __restrict__ denom,
                                 const float* __restrict__ Hb, float* __restrict__ Ob,
                                 long long total, int Npad) {
  constexpr int HC = HH * CC;
  long long i = (long long)blockIdx.x * blockDim.x + threadIdx.x;
  if (i >= total) return;
  int e = (int)(i / HC);
  int j = (int)(i - (long long)e * HC);
  int h = j / CC;
  int s = src[e], d = dst[e];
  float alpha = ebuf[(size_t)e * HH + h] / (denom[d * HH + h] + 1e-16f);
  atomicAdd(&Ob[(size_t)d * Npad + j], Hb[(size_t)s * Npad + j] * alpha);
}

// bias + ELU into next activation buffer (same leading dim Npad)
__global__ void bias_elu(const float* __restrict__ Ob, const float* __restrict__ b,
                         float* __restrict__ act, int Nn, int HC, int Npad) {
  int i = blockIdx.x * blockDim.x + threadIdx.x;
  if (i >= Nn * HC) return;
  int n = i / HC, j = i - n * HC;
  float v = Ob[(size_t)n * Npad + j] + b[j];
  v = (v > 0.f) ? v : (__expf(v) - 1.f);   // ELU alpha=1
  act[(size_t)n * Npad + j] = v;
}

// last layer: bias + log_softmax over Cc channels -> d_out [N, Cc]
__global__ void final_logsoftmax(const float* __restrict__ Ob, const float* __restrict__ b,
                                 float* __restrict__ out, int Nn, int Cc, int Npad) {
  int n = blockIdx.x * blockDim.x + threadIdx.x;
  if (n >= Nn) return;
  const float* row = Ob + (size_t)n * Npad;
  float mx = -1e30f;
  for (int c = 0; c < Cc; ++c) mx = fmaxf(mx, row[c] + b[c]);
  float se = 0.f;
  for (int c = 0; c < Cc; ++c) se += __expf(row[c] + b[c] - mx);
  float lse = mx + __logf(se);
  for (int c = 0; c < Cc; ++c) out[(size_t)n * Cc + c] = row[c] + b[c] - lse;
}

// ------------------------------- host driver -------------------------------

extern "C" void kernel_launch(void* const* d_in, const int* in_sizes, int n_in,
                              void* d_out, int out_size, void* d_ws, size_t ws_size,
                              hipStream_t stream) {
  const int N  = 50000;
  const int E  = in_sizes[1] / 2;
  const int Et = E + N;

  const float* x  = (const float*)d_in[0];
  const int*   ei = (const int*)d_in[1];

  char* ws = (char*)d_ws;
  size_t off = 0;
  auto alloc = [&](size_t bytes) -> char* {
    char* p = ws + off;
    off += (bytes + 255) & ~(size_t)255;
    return p;
  };
  int*    src   = (int*)alloc((size_t)Et * 4);
  int*    dst   = (int*)alloc((size_t)Et * 4);
  float*  act0  = (float*)alloc((size_t)N * 112 * 4);
  float*  act1  = (float*)alloc((size_t)N * 112 * 4);
  __bf16* Xbf   = (__bf16*)alloc((size_t)N * 512 * 2);
  __bf16* Wbf   = (__bf16*)alloc((size_t)512 * 112 * 2);
  float*  Hbuf  = (float*)alloc((size_t)N * 112 * 4);
  float*  Obuf  = (float*)alloc((size_t)N * 112 * 4);
  float*  es    = (float*)alloc((size_t)N * 16 * 4);
  float*  ed    = (float*)alloc((size_t)N * 16 * 4);
  float*  emax  = (float*)alloc((size_t)N * 16 * 4);
  float*  denom = (float*)alloc((size_t)N * 16 * 4);
  float*  ebuf  = (float*)alloc((size_t)Et * 10 * 4);

  dim3 blk(256);
  build_edges<<<(Et + 255) / 256, blk, 0, stream>>>(ei, E, N, src, dst);

  struct Cfg { int Fin, C, H, concat, Kpad, Npad; };
  const Cfg L[5] = {
    {512, 10, 10, 1, 512, 112},
    {100, 10,  8, 1, 128,  80},
    { 80, 10,  4, 1,  96,  48},
    { 40, 10,  1, 1,  64,  16},
    { 10, 40,  1, 0,  32,  48},
  };

  const float* actIn = x;
  int ldIn = 512;
  for (int li = 0; li < 5; ++li) {
    const Cfg& c = L[li];
    const float* W   = (const float*)d_in[2 + 4 * li];
    const float* as_ = (const float*)d_in[3 + 4 * li];
    const float* ad_ = (const float*)d_in[4 + 4 * li];
    const float* b   = (const float*)d_in[5 + 4 * li];
    const int HC = c.H * c.C;

    {
      long long tot = (long long)N * c.Kpad;
      to_bf16_pad<<<(unsigned)((tot + 255) / 256), blk, 0, stream>>>(
          actIn, ldIn, c.Fin, N, Xbf, c.Kpad);
    }
    {
      int tot = c.Kpad * c.Npad;
      w_to_bf16<<<(tot + 255) / 256, blk, 0, stream>>>(W, c.Fin, HC, Wbf, c.Kpad, c.Npad);
    }
    {
      int Mtiles = (N + 15) / 16;
      int blocks = (Mtiles + 7) / 8;         // 8 waves (one 16-row stripe each)
      int Ntiles = c.Npad / 16;
      switch (Ntiles) {
        case 7: gemm_wmma_bf16_nt<7><<<blocks, blk, 0, stream>>>(Xbf, Wbf, Hbuf, Mtiles, c.Kpad, c.Npad, N); break;
        case 5: gemm_wmma_bf16_nt<5><<<blocks, blk, 0, stream>>>(Xbf, Wbf, Hbuf, Mtiles, c.Kpad, c.Npad, N); break;
        case 3: gemm_wmma_bf16_nt<3><<<blocks, blk, 0, stream>>>(Xbf, Wbf, Hbuf, Mtiles, c.Kpad, c.Npad, N); break;
        default: gemm_wmma_bf16_nt<1><<<blocks, blk, 0, stream>>>(Xbf, Wbf, Hbuf, Mtiles, c.Kpad, c.Npad, N); break;
      }
    }
    attn_scores<<<(N * c.H + 255) / 256, blk, 0, stream>>>(
        Hbuf, c.Npad, as_, ad_, es, ed, N, c.H, c.C);
    fill_f32<<<(N * c.H + 255) / 256, blk, 0, stream>>>(emax, -1e30f, N * c.H);
    fill_f32<<<(N * c.H + 255) / 256, blk, 0, stream>>>(denom, 0.0f, N * c.H);
    {
      int tot = N * c.Npad;
      fill_f32<<<(tot + 255) / 256, blk, 0, stream>>>(Obuf, 0.0f, tot);
    }
    {
      long long tot = (long long)Et * c.H;
      edge_max<<<(unsigned)((tot + 255) / 256), blk, 0, stream>>>(
          src, dst, es, ed, ebuf, emax, Et, c.H);
      edge_exp<<<(unsigned)((tot + 255) / 256), blk, 0, stream>>>(
          dst, ebuf, emax, denom, Et, c.H);
    }
    {
      long long tot = (long long)Et * HC;
      unsigned blocks = (unsigned)((tot + 255) / 256);
      if      (li == 0) edge_aggregate_t<10, 10><<<blocks, blk, 0, stream>>>(src, dst, ebuf, denom, Hbuf, Obuf, tot, c.Npad);
      else if (li == 1) edge_aggregate_t< 8, 10><<<blocks, blk, 0, stream>>>(src, dst, ebuf, denom, Hbuf, Obuf, tot, c.Npad);
      else if (li == 2) edge_aggregate_t< 4, 10><<<blocks, blk, 0, stream>>>(src, dst, ebuf, denom, Hbuf, Obuf, tot, c.Npad);
      else if (li == 3) edge_aggregate_t< 1, 10><<<blocks, blk, 0, stream>>>(src, dst, ebuf, denom, Hbuf, Obuf, tot, c.Npad);
      else              edge_aggregate_t< 1, 40><<<blocks, blk, 0, stream>>>(src, dst, ebuf, denom, Hbuf, Obuf, tot, c.Npad);
    }

    if (li < 4) {
      float* actOut = (li & 1) ? act1 : act0;
      int tot = N * HC;
      bias_elu<<<(tot + 255) / 256, blk, 0, stream>>>(Obuf, b, actOut, N, HC, c.Npad);
      actIn = actOut;
      ldIn  = c.Npad;
    } else {
      final_logsoftmax<<<(N + 255) / 256, blk, 0, stream>>>(
          Obuf, b, (float*)d_out, N, c.C, c.Npad);
    }
  }
  (void)n_in; (void)out_size; (void)ws_size;
}
